// MultiHeadAttentionLayer_4604204941555
// MI455X (gfx1250) — matrix operand order repaired and checked by reading, not compiled, not run on hardware
//
#include <hip/hip_runtime.h>
#include <hip/hip_bf16.h>

typedef __attribute__((ext_vector_type(16))) __bf16        v16bf;
typedef __attribute__((ext_vector_type(8)))  float         v8f;
typedef __attribute__((ext_vector_type(8)))  unsigned short v8us;
typedef __attribute__((ext_vector_type(16))) unsigned short v16us;

__device__ __forceinline__ unsigned short f2bf(float f) {
    unsigned u = __float_as_uint(f);
    u += 0x7FFFu + ((u >> 16) & 1u);   // round-to-nearest-even
    return (unsigned short)(u >> 16);
}

__device__ __forceinline__ v16bf mkfrag(v8us lo, v8us hi) {
    v16us c = __builtin_shufflevector(lo, hi, 0, 1, 2, 3, 4, 5, 6, 7,
                                      8, 9, 10, 11, 12, 13, 14, 15);
    return __builtin_bit_cast(v16bf, c);
}

__device__ __forceinline__ v8f wmma_bf16(v16bf a, v16bf b, v8f c) {
    return __builtin_amdgcn_wmma_f32_16x16x32_bf16(
        false, a, false, b, (short)0, c, false, false);
}

// CDNA5 async global->LDS copy (GVS mode): per-lane LDS dest addr + per-lane
// 32-bit byte offset against a uniform 64-bit SGPR base. Tracked by ASYNCcnt.
__device__ __forceinline__ void async_ld_b128(void* lds_ptr, const void* base,
                                              unsigned byte_off) {
    unsigned lds = (unsigned)(uintptr_t)lds_ptr;  // low 32b = LDS offset
    asm volatile("global_load_async_to_lds_b128 %0, %1, %2"
                 :: "v"(lds), "v"(byte_off), "s"(base)
                 : "memory");
}
__device__ __forceinline__ void wait_async0() {
    asm volatile("s_wait_asynccnt 0x0" ::: "memory");
}

// ---------------------------------------------------------------------------
// fp32 -> bf16 bulk convert (n multiple of 4)
// ---------------------------------------------------------------------------
__global__ void f32_to_bf16_kernel(const float* __restrict__ in,
                                   unsigned short* __restrict__ out, int n) {
    int i = (blockIdx.x * blockDim.x + threadIdx.x) * 4;
    if (i >= n) return;
    float4 v = *(const float4*)(in + i);
    unsigned long long pack =
        (unsigned long long)f2bf(v.x) |
        ((unsigned long long)f2bf(v.y) << 16) |
        ((unsigned long long)f2bf(v.z) << 32) |
        ((unsigned long long)f2bf(v.w) << 48);
    *(unsigned long long*)(out + i) = pack;
}

// ---------------------------------------------------------------------------
// C = A(MxK) * W(NxK)^T + bias   (bf16 in, f32 acc, bf16 or f32 out)
// block tile 128x128, BK=32, 8 waves in 2x4, each wave 64x32 (8 WMMA accs)
// Tiles staged via global_load_async_to_lds_b128.
// ---------------------------------------------------------------------------
template <bool OUT_BF16>
__global__ __launch_bounds__(256) void gemm_bias_wmma(
    const unsigned short* __restrict__ A, const unsigned short* __restrict__ W,
    const float* __restrict__ bias, void* __restrict__ out, int M, int N,
    int K) {
    constexpr int BK = 32;
    __shared__ __attribute__((aligned(16))) unsigned short As[128][BK + 8];
    __shared__ __attribute__((aligned(16))) unsigned short Bs[128][BK + 8];

    const int tid = threadIdx.x;
    const int wave = tid >> 5, lane = tid & 31;
    const int lhalf = lane >> 4, lrow = lane & 15;
    const int wm = wave >> 2, wn = wave & 3;  // 2 x 4 wave grid
    const int m0 = blockIdx.x * 128, n0 = blockIdx.y * 128;

    v8f acc[4][2];
#pragma unroll
    for (int i = 0; i < 4; i++)
#pragma unroll
        for (int j = 0; j < 2; j++)
            acc[i][j] = (v8f){0.f, 0.f, 0.f, 0.f, 0.f, 0.f, 0.f, 0.f};

    for (int k0 = 0; k0 < K; k0 += BK) {
#pragma unroll
        for (int it = 0; it < 2; ++it) {
            int c = tid + it * 256;      // 512 chunks of 8 bf16
            int row = c >> 2;            // 0..127
            int kc = (c & 3) * 8;        // 0,8,16,24
            unsigned aoff =
                (unsigned)(((size_t)(m0 + row) * K + k0 + kc) * 2);
            unsigned boff =
                (unsigned)(((size_t)(n0 + row) * K + k0 + kc) * 2);
            async_ld_b128(&As[row][kc], A, aoff);
            async_ld_b128(&Bs[row][kc], W, boff);
        }
        wait_async0();
        __syncthreads();

        v16bf af[4], bf[2];
#pragma unroll
        for (int i = 0; i < 4; i++) {
            const unsigned short* p = &As[wm * 64 + i * 16 + lrow][0];
            af[i] = mkfrag(*(const v8us*)(p + lhalf * 8),
                           *(const v8us*)(p + 16 + lhalf * 8));
        }
#pragma unroll
        for (int j = 0; j < 2; j++) {
            const unsigned short* p = &Bs[wn * 32 + j * 16 + lrow][0];
            bf[j] = mkfrag(*(const v8us*)(p + lhalf * 8),
                           *(const v8us*)(p + 16 + lhalf * 8));
        }
#pragma unroll
        for (int i = 0; i < 4; i++)
#pragma unroll
            for (int j = 0; j < 2; j++)
                acc[i][j] = wmma_bf16(af[i], bf[j], acc[i][j]);
        __syncthreads();
    }

#pragma unroll
    for (int i = 0; i < 4; i++)
#pragma unroll
        for (int j = 0; j < 2; j++)
#pragma unroll
            for (int r = 0; r < 8; r++) {
                int m = m0 + wm * 64 + i * 16 + r + lhalf * 8;
                int n = n0 + wn * 32 + j * 16 + lrow;
                float v = acc[i][j][r] + bias[n];
                if constexpr (OUT_BF16)
                    ((unsigned short*)out)[(size_t)m * N + n] = f2bf(v);
                else
                    ((float*)out)[(size_t)m * N + n] = v;
            }
}

// ---------------------------------------------------------------------------
// Flash attention: block = 128 query rows for one (b,h); 8 waves x 16 rows.
// K tile via async-to-LDS; V staged transposed. Online softmax, fp32 accum.
// ---------------------------------------------------------------------------
__global__ __launch_bounds__(256) void flash_attn_wmma(
    const unsigned short* __restrict__ Q, const unsigned short* __restrict__ Kg,
    const unsigned short* __restrict__ Vg, unsigned short* __restrict__ Ctx) {
    constexpr int S = 2048, H = 1024, DK = 64;
    __shared__ __attribute__((aligned(16))) unsigned short Ks[32][DK + 8];
    __shared__ __attribute__((aligned(16))) unsigned short Vs[DK][32 + 8];
    __shared__ __attribute__((aligned(16))) unsigned short Ps[8][16][32 + 8];

    const int tid = threadIdx.x, wave = tid >> 5, lane = tid & 31;
    const int lhalf = lane >> 4, lrow = lane & 15;
    const int bh = blockIdx.y, b = bh >> 4, h = bh & 15;
    const int q0 = blockIdx.x * 128 + wave * 16;
    const size_t qrow = (size_t)(b * S + q0);

    // Q fragments (16 x 64 split into two 16x32 A-frags), loaded from global
    v16bf qf[2];
#pragma unroll
    for (int f = 0; f < 2; ++f) {
        const unsigned short* qp =
            Q + (qrow + lrow) * H + h * DK + f * 32 + lhalf * 8;
        qf[f] = mkfrag(*(const v8us*)qp, *(const v8us*)(qp + 16));
    }

    float mrun[8], lrun[8];
    v8f o[4];
#pragma unroll
    for (int r = 0; r < 8; r++) {
        mrun[r] = -3.0e38f;
        lrun[r] = 0.f;
    }
#pragma unroll
    for (int t = 0; t < 4; t++)
        o[t] = (v8f){0.f, 0.f, 0.f, 0.f, 0.f, 0.f, 0.f, 0.f};

    for (int kv0 = 0; kv0 < S; kv0 += 32) {
        {   // K tile async to LDS; V tile through regs (transposed store)
            int row = tid >> 3;        // 0..31 (kv index)
            int dc = (tid & 7) * 8;    // 0..56
            unsigned koff = (unsigned)(((size_t)(b * S + kv0 + row) * H +
                                        h * DK + dc) * 2);
            async_ld_b128(&Ks[row][dc], Kg, koff);
            const unsigned short* vp =
                Vg + (size_t)(b * S + kv0 + row) * H + h * DK + dc;
            v8us vv = *(const v8us*)vp;
#pragma unroll
            for (int e = 0; e < 8; e++) Vs[dc + e][row] = vv[e];
        }
        wait_async0();
        __syncthreads();

        // scores S = Q * K^T  (two 16x16 tiles over 32 keys)
        v8f sc[2];
        sc[0] = (v8f){0.f, 0.f, 0.f, 0.f, 0.f, 0.f, 0.f, 0.f};
        sc[1] = (v8f){0.f, 0.f, 0.f, 0.f, 0.f, 0.f, 0.f, 0.f};
#pragma unroll
        for (int j = 0; j < 2; j++)
#pragma unroll
            for (int f = 0; f < 2; f++) {
                const unsigned short* p =
                    &Ks[j * 16 + lrow][f * 32 + lhalf * 8];
                v16bf kb = mkfrag(*(const v8us*)p, *(const v8us*)(p + 16));
                sc[j] = wmma_bf16(qf[f], kb, sc[j]);
            }

        // online softmax (row spans 16 lanes of one VGPR -> 16-lane xor tree)
        float pj[2][8];
#pragma unroll
        for (int r = 0; r < 8; r++) {
            float s0 = sc[0][r] * 0.125f, s1 = sc[1][r] * 0.125f;
            float tmax = fmaxf(s0, s1);
#pragma unroll
            for (int off = 1; off < 16; off <<= 1)
                tmax = fmaxf(tmax, __shfl_xor(tmax, off, 32));
            float mnew = fmaxf(mrun[r], tmax);
            float corr = __expf(mrun[r] - mnew);
            float p0 = __expf(s0 - mnew);
            float p1 = __expf(s1 - mnew);
            float psum = p0 + p1;
#pragma unroll
            for (int off = 1; off < 16; off <<= 1)
                psum += __shfl_xor(psum, off, 32);
            lrun[r] = lrun[r] * corr + psum;
            mrun[r] = mnew;
#pragma unroll
            for (int t = 0; t < 4; t++) o[t][r] = o[t][r] * corr;
            pj[0][r] = p0;
            pj[1][r] = p1;
        }

        // P (C-layout) -> LDS -> A-fragment layout, bf16
#pragma unroll
        for (int j = 0; j < 2; j++)
#pragma unroll
            for (int r = 0; r < 8; r++)
                Ps[wave][r + lhalf * 8][j * 16 + lrow] = f2bf(pj[j][r]);
        asm volatile("s_wait_dscnt 0x0" ::: "memory");

        const unsigned short* pp = &Ps[wave][lrow][0];
        v16bf pf = mkfrag(*(const v8us*)(pp + lhalf * 8),
                          *(const v8us*)(pp + 16 + lhalf * 8));

        // ctx += P * V   (four 16-wide d tiles, K=32 in one WMMA each)
#pragma unroll
        for (int t = 0; t < 4; t++) {
            const unsigned short* vptr = &Vs[t * 16 + lrow][lhalf * 8];
            v16bf vf = mkfrag(*(const v8us*)vptr, *(const v8us*)(vptr + 16));
            o[t] = wmma_bf16(pf, vf, o[t]);
        }
        __syncthreads();
    }

    // normalize and store ctx as bf16 in (B,S,H) layout
    float linv[8];
#pragma unroll
    for (int r = 0; r < 8; r++) linv[r] = 1.0f / lrun[r];
#pragma unroll
    for (int t = 0; t < 4; t++)
#pragma unroll
        for (int r = 0; r < 8; r++) {
            float v = o[t][r] * linv[r];
            Ctx[(qrow + r + lhalf * 8) * H + h * DK + t * 16 + lrow] = f2bf(v);
        }
}

// ---------------------------------------------------------------------------
extern "C" void kernel_launch(void* const* d_in, const int* in_sizes, int n_in,
                              void* d_out, int out_size, void* d_ws,
                              size_t ws_size, hipStream_t stream) {
    const float* X  = (const float*)d_in[0];
    const float* Wq = (const float*)d_in[1];
    const float* bq = (const float*)d_in[2];
    const float* Wk = (const float*)d_in[3];
    const float* bk = (const float*)d_in[4];
    const float* Wv = (const float*)d_in[5];
    const float* bv = (const float*)d_in[6];
    const float* Wo = (const float*)d_in[7];
    const float* bo = (const float*)d_in[8];

    const int B = 4, S = 2048, D = 1024, H = 1024;
    const int M = B * S;  // 8192

    unsigned short* ws  = (unsigned short*)d_ws;
    unsigned short* Xb  = ws;                        // M*D
    unsigned short* Wqb = Xb + (size_t)M * D;        // H*D
    unsigned short* Wkb = Wqb + (size_t)H * D;
    unsigned short* Wvb = Wkb + (size_t)H * D;
    unsigned short* Wob = Wvb + (size_t)H * D;
    unsigned short* Qb  = Wob + (size_t)H * H;       // M*H
    unsigned short* Kb  = Qb + (size_t)M * H;
    unsigned short* Vb  = Kb + (size_t)M * H;
    unsigned short* Cb  = Vb + (size_t)M * H;

    // fp32 -> bf16 conversions
    {
        int n = M * D;
        f32_to_bf16_kernel<<<(n / 4 + 255) / 256, 256, 0, stream>>>(X, Xb, n);
        n = H * D;
        f32_to_bf16_kernel<<<(n / 4 + 255) / 256, 256, 0, stream>>>(Wq, Wqb, n);
        f32_to_bf16_kernel<<<(n / 4 + 255) / 256, 256, 0, stream>>>(Wk, Wkb, n);
        f32_to_bf16_kernel<<<(n / 4 + 255) / 256, 256, 0, stream>>>(Wv, Wvb, n);
        f32_to_bf16_kernel<<<(n / 4 + 255) / 256, 256, 0, stream>>>(Wo, Wob, n);
    }

    dim3 ggrid(M / 128, H / 128);  // (64, 8)
    gemm_bias_wmma<true><<<ggrid, 256, 0, stream>>>(Xb, Wqb, bq, Qb, M, H, D);
    gemm_bias_wmma<true><<<ggrid, 256, 0, stream>>>(Xb, Wkb, bk, Kb, M, H, D);
    gemm_bias_wmma<true><<<ggrid, 256, 0, stream>>>(Xb, Wvb, bv, Vb, M, H, D);

    flash_attn_wmma<<<dim3(S / 128, B * 16), 256, 0, stream>>>(Qb, Kb, Vb, Cb);

    gemm_bias_wmma<false><<<ggrid, 256, 0, stream>>>(Cb, Wob, bo, d_out, M, H,
                                                     H);
}
